// RGN_35888746725974
// MI455X (gfx1250) — matrix operand order repaired and checked by reading, not compiled, not run on hardware
//
#include <hip/hip_runtime.h>
#include <hip/hip_bf16.h>

// ---- problem constants -------------------------------------------------
#define HH   800            // hidden
#define DD   42             // input features
#define AA   20             // alphabet
#define LLEN 700            // sequence length
#define BB   32             // batch
#define MROWS (LLEN * BB)   // 22400, multiple of 32
#define KCHUNK 512          // K-slice staged in LDS by the TDM

typedef __attribute__((ext_vector_type(16))) _Float16 v16h;
typedef __attribute__((ext_vector_type(8)))  _Float16 v8h;
typedef __attribute__((ext_vector_type(8)))  float    v8f;
typedef __attribute__((ext_vector_type(4)))  unsigned int u32x4;
typedef __attribute__((ext_vector_type(8)))  int      i32x8;
typedef __attribute__((ext_vector_type(4)))  int      i32x4;

union AOp { v16h v; v8h h[2]; };

// =======================================================================
// Tensor Data Mover: stage a [rows x kc] f16 tile (row stride ld elems)
// into LDS at byte-offset ldsOff, padding 4 dwords after each row so the
// 16 A-operand readers hit 16 distinct LDS banks. kc must be a power of 2
// (>= 16). Issued once per wave; caller guards to one wave per block.
// =======================================================================
__device__ inline void tdm_load_tile_2d(unsigned ldsOff, const void* gaddr,
                                        int kc, int rows, int ld) {
    const unsigned long long ga = (unsigned long long)(uintptr_t)gaddr;
    const unsigned padIntervalCode =
        (unsigned)(31 - __builtin_clz((unsigned)kc)) - 2u;  // 2^c * 8B == kc*2B
    u32x4 g0 = { 1u,                                   // count=1 (user D#)
                 ldsOff,                               // lds_addr
                 (unsigned)ga,                         // global_addr[31:0]
                 (unsigned)((ga >> 32) & 0x01ffffffu)  // global_addr[56:32]
                     | (2u << 30) };                   // type=2 ("image")
    const unsigned d0 = (1u << 16)                     // data_size = 2 bytes
                      | (1u << 20)                     // pad_enable
                      | (padIntervalCode << 22)        // pad every row
                      | (3u << 25);                    // pad_amount = 4 dwords
    i32x8 g1 = { (int)d0,
                 (int)(((unsigned)kc & 0xffffu) << 16),                  // tensor_dim0 lo
                 (int)((((unsigned)kc >> 16) & 0xffffu)
                     | (((unsigned)rows & 0xffffu) << 16)),              // td0 hi | td1 lo
                 (int)((((unsigned)rows >> 16) & 0xffffu)
                     | (((unsigned)kc & 0xffffu) << 16)),                // td1 hi | tile_dim0
                 (int)((unsigned)rows & 0xffffu),                        // tile_dim1 (tile_dim2=0)
                 (int)ld,                                                // tensor_dim0_stride lo32
                 0, 0 };
    i32x4 z4 = {0, 0, 0, 0};
#if __has_builtin(__builtin_amdgcn_tensor_load_to_lds)
# if defined(__clang_major__) && __clang_major__ >= 23
    i32x8 z8 = {0, 0, 0, 0, 0, 0, 0, 0};
    __builtin_amdgcn_tensor_load_to_lds(g0, g1, z4, z4, z8, 0);
# else
    __builtin_amdgcn_tensor_load_to_lds(g0, g1, z4, z4, 0);
# endif
    __builtin_amdgcn_s_wait_tensorcnt(0);
#endif
}

// =======================================================================
// f32 -> f16 convert with zero padding: dst[Rd][Cd] <- src[Rs][Cs]
// =======================================================================
__global__ void convert_pad_f16(const float* __restrict__ src,
                                _Float16* __restrict__ dst,
                                int Rs, int Cs, int Rd, int Cd) {
    size_t i = (size_t)blockIdx.x * blockDim.x + threadIdx.x;
    size_t tot = (size_t)Rd * Cd;
    if (i >= tot) return;
    int r = (int)(i / Cd), c = (int)(i % Cd);
    dst[i] = (r < Rs && c < Cs) ? (_Float16)src[(size_t)r * Cs + c] : (_Float16)0.f;
}

// x: [B][L][D] f32  ->  x16: [L][B][64] f16 (K padded 42->64, L/B transposed)
__global__ void pack_x_f16(const float* __restrict__ x, _Float16* __restrict__ dst) {
    size_t i = (size_t)blockIdx.x * blockDim.x + threadIdx.x;
    if (i >= (size_t)MROWS * 64) return;
    int c  = (int)(i & 63);
    int lb = (int)(i >> 6);
    int b  = lb % BB, l = lb / BB;
    dst[i] = (c < DD) ? (_Float16)x[((size_t)b * LLEN + l) * DD + c] : (_Float16)0.f;
}

// =======================================================================
// WMMA GEMM:  C[M,N] = A[M,K](f16,row-major,lda=K) x B[N,K](f16,row-major)^T
// Block (32,4): block owns 32 rows x 128 cols. The 32xKCHUNK A panel is
// staged into LDS by the TDM (shared by all 4 waves); each wave register-
// blocks 2M x 2N tiles (4 WMMAs per fetch group -> ~33 FLOP/B from L2).
// K must be a multiple of 32 and chunk into power-of-2 slices (<=KCHUNK).
// =======================================================================
__global__ __launch_bounds__(128) void gemm_f16_wmma(
    const _Float16* __restrict__ Amat,
    const _Float16* __restrict__ Bmat, int ldb,
    const float* __restrict__ bias,
    _Float16* __restrict__ Ch, float* __restrict__ Cf, int ldc,
    int N, int K, int Nvalid)
{
    extern __shared__ _Float16 smemA[];        // [32][kc + 8] per chunk
    const int lane = threadIdx.x;
    const int m0 = blockIdx.x * 32;
    const int n0 = blockIdx.y * 128 + threadIdx.y * 32;
    const bool active = (n0 < N);
    const int nn = lane & 15, hi = lane >> 4;

    v8f acc[2][2] = {};
    for (int kb = 0; kb < K; kb += KCHUNK) {
        const int kc = (K - kb < KCHUNK) ? (K - kb) : KCHUNK;
        const int lstride = kc + 8;            // f16 elems (bank-conflict pad)
        __syncthreads();                       // previous chunk fully consumed
        if (threadIdx.y == 0)                  // one TDM issue per block
            tdm_load_tile_2d((unsigned)(uintptr_t)smemA,
                             Amat + (size_t)m0 * K + kb, kc, 32, K);
        __syncthreads();                       // panel visible to all waves
        if (!active) continue;

        for (int k0 = 0; k0 < kc; k0 += 32) {
            AOp a[2];
            #pragma unroll
            for (int mt = 0; mt < 2; ++mt) {   // A from LDS (ds_load_b128)
                const _Float16* ap = smemA + (mt * 16 + nn) * lstride + k0 + hi * 16;
                a[mt].h[0] = *(const v8h*)ap;
                a[mt].h[1] = *(const v8h*)(ap + 8);
            }
            #pragma unroll
            for (int nt = 0; nt < 2; ++nt) {   // B streamed from L2
                const _Float16* bp = Bmat + (size_t)(n0 + nt * 16 + nn) * ldb
                                          + kb + k0 + hi * 16;
                v16h b = *(const v16h*)bp;
                #pragma unroll
                for (int mt = 0; mt < 2; ++mt)
                    acc[mt][nt] = __builtin_amdgcn_wmma_f32_16x16x32_f16(
                        false, a[mt].v, false, b, (short)0, acc[mt][nt], false, false);
            }
        }
    }
    if (!active) return;

    #pragma unroll
    for (int nt = 0; nt < 2; ++nt) {
        const int n = n0 + nt * 16 + nn;
        const float bv = (bias && n < Nvalid) ? bias[n] : 0.f;
        #pragma unroll
        for (int mt = 0; mt < 2; ++mt) {
            #pragma unroll
            for (int r = 0; r < 8; ++r) {
                const size_t row = (size_t)(m0 + mt * 16 + r + 8 * hi);
                if (Ch)             Ch[row * ldc + n] = (_Float16)(acc[mt][nt][r] + bv);
                else if (n < Nvalid) Cf[row * ldc + n] = acc[mt][nt][r] + bv;
            }
        }
    }
}

// =======================================================================
// One bidirectional LSTM time step (both directions via gridDim.z).
//   gates = pre[t] + h_prev @ Whh^T ; cell update ; h -> hstore (f16)
// grid (25, 2, 2), block (32,2): wave owns a 16-wide hidden chunk x 4 gates
// (one A fetch feeds 4 WMMAs). Whh is read twice per step (once per 16-row batch tile).
// =======================================================================
__global__ __launch_bounds__(64) void lstm_step_kernel(
    _Float16* __restrict__ hstore,
    const _Float16* __restrict__ preF, const _Float16* __restrict__ preR,
    const _Float16* __restrict__ WhhF, const _Float16* __restrict__ WhhR,
    float* __restrict__ cbuf,          // [2][B][H] f32
    int s)
{
    const int lane  = threadIdx.x;
    const int chunk = blockIdx.x * 2 + threadIdx.y;    // 0..49
    const int m0    = blockIdx.y * 16;                 // batch tile
    const int dir   = blockIdx.z;
    const int t     = dir ? (LLEN - 1 - s) : s;
    const int tprev = dir ? (t + 1) : (t - 1);
    const _Float16* pre = dir ? preR : preF;
    const _Float16* Whh = dir ? WhhR : WhhF;

    const int nn = lane & 15, hi = lane >> 4;
    const int jcol = chunk * 16;

    v8f acc[4];
    #pragma unroll
    for (int g = 0; g < 4; ++g) {
        const _Float16* p = pre + (size_t)t * BB * 3200 + g * HH + jcol + nn;
        #pragma unroll
        for (int r = 0; r < 8; ++r)
            acc[g][r] = (float)p[(size_t)(m0 + r + 8 * hi) * 3200];
    }

    if (s > 0) {
        const _Float16* hprev = hstore + (size_t)tprev * BB * 1600 + dir * HH;
        for (int k0 = 0; k0 < HH; k0 += 32) {
            AOp a;
            const _Float16* ap = hprev + (size_t)(m0 + nn) * 1600 + k0 + hi * 16;
            a.h[0] = *(const v8h*)ap;
            a.h[1] = *(const v8h*)(ap + 8);
            #pragma unroll
            for (int g = 0; g < 4; ++g) {
                const _Float16* bp = Whh + (size_t)(g * HH + jcol + nn) * HH
                                         + k0 + hi * 16;
                v16h b = *(const v16h*)bp;
                acc[g] = __builtin_amdgcn_wmma_f32_16x16x32_f16(
                             false, a.v, false, b, (short)0, acc[g], false, false);
            }
        }
    }

    float* cp = cbuf + (size_t)dir * BB * HH;
    _Float16* hout = hstore + (size_t)t * BB * 1600 + dir * HH;
    #pragma unroll
    for (int r = 0; r < 8; ++r) {
        const int m = m0 + r + 8 * hi;
        const int n = jcol + nn;
        const float ig = 1.f / (1.f + __expf(-acc[0][r]));
        const float fg = 1.f / (1.f + __expf(-acc[1][r]));
        const float gg = tanhf(acc[2][r]);
        const float og = 1.f / (1.f + __expf(-acc[3][r]));
        const float cold = (s > 0) ? cp[(size_t)m * HH + n] : 0.f;
        const float cnew = fg * cold + ig * gg;
        cp[(size_t)m * HH + n] = cnew;
        hout[(size_t)m * 1600 + n] = (_Float16)(og * tanhf(cnew));
    }
}

// =======================================================================
// Softmax over the BATCH dim (dim=1): one wave per (l, a); lane = batch.
// =======================================================================
__global__ __launch_bounds__(128) void softmax_batch(float* __restrict__ logits) {
    const int id = blockIdx.x * blockDim.y + threadIdx.y;
    if (id >= LLEN * AA) return;
    const int l = id / AA, a = id % AA, b = threadIdx.x;
    const size_t ix = ((size_t)l * BB + b) * AA + a;
    float v = logits[ix];
    float mx = v;
    #pragma unroll
    for (int o = 16; o; o >>= 1) mx = fmaxf(mx, __shfl_xor(mx, o, 32));
    const float e = __expf(v - mx);
    float sm = e;
    #pragma unroll
    for (int o = 16; o; o >>= 1) sm += __shfl_xor(sm, o, 32);
    logits[ix] = e / sm;
}

// probs [L*B][A] x {sin,cos}(alphabet[A][3]) -> dih = atan2 : [L*B][3]
__global__ void dihedral_kernel(const float* __restrict__ probs,
                                const float* __restrict__ alphabet,
                                float* __restrict__ dih) {
    const int i = blockIdx.x * blockDim.x + threadIdx.x;
    if (i >= LLEN * BB) return;
    float s[3] = {0.f, 0.f, 0.f}, c[3] = {0.f, 0.f, 0.f};
    for (int a = 0; a < AA; ++a) {
        const float p = probs[(size_t)i * AA + a];
        #pragma unroll
        for (int k = 0; k < 3; ++k) {
            const float ang = alphabet[a * 3 + k];
            s[k] += p * __sinf(ang);
            c[k] += p * __cosf(ang);
        }
    }
    #pragma unroll
    for (int k = 0; k < 3; ++k) dih[(size_t)i * 3 + k] = atan2f(s[k], c[k]);
}

// Sequential NeRF extension: 32 lanes = 32 independent batch chains.
__global__ void nerf_kernel(const float* __restrict__ dih, float* __restrict__ out) {
    const int b = threadIdx.x;
    float ax = -0.70710678f, ay = 1.22474487f, az = 0.f;
    float bx = -1.41421356f, by = 0.f,         bz = 0.f;
    float cx = 0.f,          cy = 0.f,         cz = 0.f;
    const float BLEN[3] = {145.801f, 152.326f, 132.868f};
    const float BANG[3] = {2.124f, 1.941f, 2.028f};
    float rc[3], rs[3];
    #pragma unroll
    for (int j = 0; j < 3; ++j) {
        rc[j] = BLEN[j] * __cosf(3.14159265358979f - BANG[j]);
        rs[j] = BLEN[j] * __sinf(3.14159265358979f - BANG[j]);
    }
    for (int i = 0; i < 3 * LLEN; ++i) {
        const int j = i % 3, l = i / 3;
        const float d = dih[((size_t)l * BB + b) * 3 + j];
        const float px = rc[j], py = __cosf(d) * rs[j], pz = __sinf(d) * rs[j];
        float ux = cx - bx, uy = cy - by, uz = cz - bz;
        float inv = rsqrtf(ux * ux + uy * uy + uz * uz + 1e-12f);
        ux *= inv; uy *= inv; uz *= inv;
        const float vx = bx - ax, vy = by - ay, vz = bz - az;
        float nx = vy * uz - vz * uy, ny = vz * ux - vx * uz, nz = vx * uy - vy * ux;
        inv = rsqrtf(nx * nx + ny * ny + nz * nz + 1e-12f);
        nx *= inv; ny *= inv; nz *= inv;
        const float mx = ny * uz - nz * uy, my = nz * ux - nx * uz, mz = nx * uy - ny * ux;
        const float ox = cx + ux * px + mx * py + nx * pz;
        const float oy = cy + uy * px + my * py + ny * pz;
        const float oz = cz + uz * px + mz * py + nz * pz;
        const size_t o = ((size_t)i * BB + b) * 3;
        out[o] = ox; out[o + 1] = oy; out[o + 2] = oz;
        ax = bx; ay = by; az = bz;
        bx = cx; by = cy; bz = cz;
        cx = ox; cy = oy; cz = oz;
    }
}

// =======================================================================
// host orchestration
// =======================================================================
extern "C" void kernel_launch(void* const* d_in, const int* in_sizes, int n_in,
                              void* d_out, int out_size, void* d_ws, size_t ws_size,
                              hipStream_t stream) {
    const float* x     = (const float*)d_in[0];
    const float* Wih0f = (const float*)d_in[1];
    const float* Whh0f = (const float*)d_in[2];
    const float* b0f   = (const float*)d_in[3];
    const float* Wih0r = (const float*)d_in[4];
    const float* Whh0r = (const float*)d_in[5];
    const float* b0r   = (const float*)d_in[6];
    const float* Wih1f = (const float*)d_in[7];
    const float* Whh1f = (const float*)d_in[8];
    const float* b1f   = (const float*)d_in[9];
    const float* Wih1r = (const float*)d_in[10];
    const float* Whh1r = (const float*)d_in[11];
    const float* b1r   = (const float*)d_in[12];
    const float* Wlin  = (const float*)d_in[13];
    const float* blin  = (const float*)d_in[14];
    const float* alpha = (const float*)d_in[15];

    size_t off = 0;
    auto carve = [&](size_t bytes) -> char* {
        char* p = (char*)d_ws + off;
        off += (bytes + 255) & ~(size_t)255;
        return p;
    };
    _Float16* preF    = (_Float16*)carve((size_t)MROWS * 3200 * 2);
    _Float16* preR    = (_Float16*)carve((size_t)MROWS * 3200 * 2);
    _Float16* h1f16   = (_Float16*)carve((size_t)MROWS * 1600 * 2);
    _Float16* outf16  = (_Float16*)carve((size_t)MROWS * 1600 * 2);
    _Float16* Whh0f16 = (_Float16*)carve((size_t)3200 * HH * 2);
    _Float16* Whh0r16 = (_Float16*)carve((size_t)3200 * HH * 2);
    _Float16* Whh1f16 = (_Float16*)carve((size_t)3200 * HH * 2);
    _Float16* Whh1r16 = (_Float16*)carve((size_t)3200 * HH * 2);
    _Float16* Wih0f16 = (_Float16*)carve((size_t)3200 * 64 * 2);
    _Float16* Wih0r16 = (_Float16*)carve((size_t)3200 * 64 * 2);
    _Float16* Wih1f16 = (_Float16*)carve((size_t)3200 * 1600 * 2);
    _Float16* Wih1r16 = (_Float16*)carve((size_t)3200 * 1600 * 2);
    _Float16* Wlin16  = (_Float16*)carve((size_t)32 * 1600 * 2);
    _Float16* x16     = (_Float16*)carve((size_t)MROWS * 64 * 2);
    float*    cbuf    = (float*)   carve((size_t)2 * BB * HH * 4);
    float*    logits  = (float*)   carve((size_t)MROWS * AA * 4);
    float*    dih     = (float*)   carve((size_t)MROWS * 3 * 4);
    (void)ws_size; (void)in_sizes; (void)n_in; (void)out_size;

    auto cvGrid = [](size_t n) { return dim3((unsigned)((n + 255) / 256)); };
    auto ldsFor = [](int K) {
        int kc = (K < KCHUNK) ? K : KCHUNK;
        return (size_t)(32 * (kc + 8) + 16) * 2;   // padded A panel
    };

    // ---- 1. weight / input casts to f16 (stay L2-resident) ------------
    convert_pad_f16<<<cvGrid((size_t)3200 * HH), 256, 0, stream>>>(Whh0f, Whh0f16, 3200, HH, 3200, HH);
    convert_pad_f16<<<cvGrid((size_t)3200 * HH), 256, 0, stream>>>(Whh0r, Whh0r16, 3200, HH, 3200, HH);
    convert_pad_f16<<<cvGrid((size_t)3200 * HH), 256, 0, stream>>>(Whh1f, Whh1f16, 3200, HH, 3200, HH);
    convert_pad_f16<<<cvGrid((size_t)3200 * HH), 256, 0, stream>>>(Whh1r, Whh1r16, 3200, HH, 3200, HH);
    convert_pad_f16<<<cvGrid((size_t)3200 * 64), 256, 0, stream>>>(Wih0f, Wih0f16, 3200, DD, 3200, 64);
    convert_pad_f16<<<cvGrid((size_t)3200 * 64), 256, 0, stream>>>(Wih0r, Wih0r16, 3200, DD, 3200, 64);
    convert_pad_f16<<<cvGrid((size_t)3200 * 1600), 256, 0, stream>>>(Wih1f, Wih1f16, 3200, 1600, 3200, 1600);
    convert_pad_f16<<<cvGrid((size_t)3200 * 1600), 256, 0, stream>>>(Wih1r, Wih1r16, 3200, 1600, 3200, 1600);
    convert_pad_f16<<<cvGrid((size_t)32 * 1600), 256, 0, stream>>>(Wlin, Wlin16, AA, 1600, 32, 1600);
    pack_x_f16<<<cvGrid((size_t)MROWS * 64), 256, 0, stream>>>(x, x16);

    const dim3 gemmBlk(32, 4);

    // ---- 2. layer-0 input projections (K=64) --------------------------
    gemm_f16_wmma<<<dim3(MROWS / 32, 3200 / 128), gemmBlk, ldsFor(64), stream>>>(
        x16, Wih0f16, 64, b0f, preF, nullptr, 3200, 3200, 64, 3200);
    gemm_f16_wmma<<<dim3(MROWS / 32, 3200 / 128), gemmBlk, ldsFor(64), stream>>>(
        x16, Wih0r16, 64, b0r, preR, nullptr, 3200, 3200, 64, 3200);

    // ---- 3. layer-0 recurrence (700 steps, fwd+rev concurrent) --------
    for (int s = 0; s < LLEN; ++s)
        lstm_step_kernel<<<dim3(25, 2, 2), dim3(32, 2), 0, stream>>>(
            h1f16, preF, preR, Whh0f16, Whh0r16, cbuf, s);

    // ---- 4. layer-1 input projections (the big parallel GEMM, K=1600) -
    gemm_f16_wmma<<<dim3(MROWS / 32, 3200 / 128), gemmBlk, ldsFor(1600), stream>>>(
        h1f16, Wih1f16, 1600, b1f, preF, nullptr, 3200, 3200, 1600, 3200);
    gemm_f16_wmma<<<dim3(MROWS / 32, 3200 / 128), gemmBlk, ldsFor(1600), stream>>>(
        h1f16, Wih1r16, 1600, b1r, preR, nullptr, 3200, 3200, 1600, 3200);

    // ---- 5. layer-1 recurrence ----------------------------------------
    for (int s = 0; s < LLEN; ++s)
        lstm_step_kernel<<<dim3(25, 2, 2), dim3(32, 2), 0, stream>>>(
            outf16, preF, preR, Whh1f16, Whh1r16, cbuf, s);

    // ---- 6. linear head (N padded 20->32, f32 out) --------------------
    gemm_f16_wmma<<<dim3(MROWS / 32, 1), gemmBlk, ldsFor(1600), stream>>>(
        outf16, Wlin16, 1600, blin, nullptr, logits, AA, 32, 1600, AA);

    // ---- 7. softmax over batch, angles, NeRF --------------------------
    softmax_batch<<<dim3((LLEN * AA + 3) / 4), dim3(32, 4), 0, stream>>>(logits);
    dihedral_kernel<<<cvGrid((size_t)LLEN * BB), 256, 0, stream>>>(logits, alpha, dih);
    nerf_kernel<<<1, 32, 0, stream>>>(dih, (float*)d_out);
}
